// InfoNCE_1211180777816
// MI455X (gfx1250) — matrix operand order repaired and checked by reading, not compiled, not run on hardware
//
#include <hip/hip_runtime.h>
#include <hip/hip_bf16.h>

typedef __attribute__((ext_vector_type(16))) _Float16 v16h;
typedef __attribute__((ext_vector_type(8)))  _Float16 v8h;
typedef __attribute__((ext_vector_type(8)))  float    v8f;
typedef __attribute__((ext_vector_type(4)))  float    v4f;

#define B_ROWS 4096
#define D_DIM  512
#define NROWS  8192                      // 2B
#define KCHUNKS 16                       // 512 / 32
#define JSEG    8                        // column segments (determinism, occupancy)
#define JSEG_COLS (NROWS / JSEG)         // 1024
#define JTILES  (JSEG_COLS / 16)         // 64 j-tiles per segment
#define WAVES_PER_BLOCK 8
#define ROW_BLOCKS (NROWS / (WAVES_PER_BLOCK * 16))   // 64

// LDS B-tile: 16 rows, padded row stride 520 halves (1040 B) -> lane l15 starts
// at bank 4*l15; 16 lanes x b128 tile all 64 banks conflict-free.
#define LDS_ROW_H 520
#define LDS_TILE_H (16 * LDS_ROW_H)      // halves per buffer

// exp(x/T) = exp2(x * (1/T)/ln2), T = 0.5
#define INV_T_LOG2E 2.885390081777927f
#define LN2F 0.6931471805599453f

// ---------------------------------------------------------------------------
// Kernel 1: L2-normalize rows of concat(poly, c) (fp32) -> f16 Z in workspace.
// One wave per row; lane handles 16 consecutive floats.
// ---------------------------------------------------------------------------
__global__ void infonce_normalize(const float* __restrict__ poly,
                                  const float* __restrict__ cemb,
                                  _Float16* __restrict__ zh) {
    const int wave = threadIdx.x >> 5;
    const int lane = threadIdx.x & 31;
    const int row  = blockIdx.x * WAVES_PER_BLOCK + wave;

    const float* src = (row < B_ROWS) ? (poly + (size_t)row * D_DIM)
                                      : (cemb + (size_t)(row - B_ROWS) * D_DIM);
    const v4f* s4 = (const v4f*)(src + lane * 16);
    v4f a0 = s4[0], a1 = s4[1], a2 = s4[2], a3 = s4[3];

    float ss = 0.f;
#pragma unroll
    for (int k = 0; k < 4; ++k)
        ss += a0[k]*a0[k] + a1[k]*a1[k] + a2[k]*a2[k] + a3[k]*a3[k];
    ss += __shfl_xor(ss, 1, 32);
    ss += __shfl_xor(ss, 2, 32);
    ss += __shfl_xor(ss, 4, 32);
    ss += __shfl_xor(ss, 8, 32);
    ss += __shfl_xor(ss, 16, 32);

    const float scale = 1.0f / fmaxf(sqrtf(ss), 1e-12f);

    v8h h0, h1;
#pragma unroll
    for (int k = 0; k < 4; ++k) {
        h0[k]     = (_Float16)(a0[k] * scale);
        h0[4 + k] = (_Float16)(a1[k] * scale);
        h1[k]     = (_Float16)(a2[k] * scale);
        h1[4 + k] = (_Float16)(a3[k] * scale);
    }
    v8h* dst = (v8h*)(zh + (size_t)row * D_DIM + lane * 16);
    dst[0] = h0;
    dst[1] = h1;
}

// ---------------------------------------------------------------------------
// Kernel 2: Gram tiles via v_wmma_f32_16x16x32_f16.
// A fragments (16-row stripe, K=512) resident in VGPRs.
// B tiles: async global->LDS double buffer (ASYNCcnt), consumed via ds_load.
//   - tile jt+1 is prefetched with global_load_async_to_lds_b128 while tile jt
//     feeds the WMMAs; s_wait_asynccnt 0 + barrier closes each iteration.
//   - one copy per BLOCK (8 waves share it) instead of 8 redundant streams.
// ---------------------------------------------------------------------------
__global__ void __launch_bounds__(256, 1)
infonce_gram(const _Float16* __restrict__ zh,
             float* __restrict__ rowsum_part,  // [JSEG][NROWS]
             float* __restrict__ pos) {        // [NROWS]
    __shared__ _Float16 btile[2][LDS_TILE_H];

    const int wave = threadIdx.x >> 5;
    const int lane = threadIdx.x & 31;
    const int hi   = lane >> 4;       // half-wave: 0 or 1
    const int l15  = lane & 15;
    const int seg  = blockIdx.y;
    const int i0   = (blockIdx.x * WAVES_PER_BLOCK + wave) * 16;
    const int jbase = seg * JSEG_COLS;

    // ---- A fragments: row i0+l15; ISA 16-bit A 16x32 layout:
    // lane<16:  halves 0..7 = K c*32+0..7,   halves 8..15 = K c*32+16..23
    // lane>=16: halves 0..7 = K c*32+8..15,  halves 8..15 = K c*32+24..31
    const _Float16* arow = zh + (size_t)(i0 + l15) * D_DIM + hi * 8;
    v16h afrag[KCHUNKS];
#pragma unroll
    for (int c = 0; c < KCHUNKS; ++c) {
        const v8h* p = (const v8h*)(arow + c * 32);
        v16h a;
        ((v8h*)&a)[0] = p[0];   // K chunk base + {0..7 | 8..15}
        ((v8h*)&a)[1] = p[2];   // +16 halves  + {16..23 | 24..31}
        afrag[c] = a;
    }

    // Async-copy addressing: this wave moves rows 2w (lanes 0-15) and 2w+1
    // (lanes 16-31) of each 16-row tile; 4 issues x 16B/lane cover 1024B/row.
    const int  crow     = 2 * wave + hi;                     // tile row this half-wave copies
    const uint32_t lds0 = (uint32_t)(uintptr_t)&btile[0][0]; // LDS byte offset of buffers
    const uint32_t ldsrow = lds0 + (uint32_t)crow * (LDS_ROW_H * 2) + (uint32_t)l15 * 16;
    const uint32_t ldsbufbytes = LDS_TILE_H * 2;

    float rowpart[8];
#pragma unroll
    for (int r = 0; r < 8; ++r) rowpart[r] = 0.f;

    const int jpos = i0 ^ B_ROWS;     // tile whose diagonal holds the positives

    // Prologue: fetch tile 0 into buffer 0.
    {
        const _Float16* g = zh + (size_t)(jbase + crow) * D_DIM + l15 * 8;
#pragma unroll
        for (int i = 0; i < 4; ++i) {
            asm volatile("global_load_async_to_lds_b128 %0, %1, off offset:%c2"
                         :: "v"(ldsrow), "v"(g), "n"(i * 256) : "memory");
        }
        asm volatile("s_wait_asynccnt 0x0" ::: "memory");
    }
    __syncthreads();

    for (int jt = 0; jt < JTILES; ++jt) {
        const int j0 = jbase + jt * 16;

        // Prefetch next tile into the other buffer (overlaps with WMMAs below).
        if (jt + 1 < JTILES) {
            const uint32_t ldsdst = ldsrow + ((jt + 1) & 1) * ldsbufbytes;
            const _Float16* g = zh + (size_t)(j0 + 16 + crow) * D_DIM + l15 * 8;
#pragma unroll
            for (int i = 0; i < 4; ++i) {
                asm volatile("global_load_async_to_lds_b128 %0, %1, off offset:%c2"
                             :: "v"(ldsdst), "v"(g), "n"(i * 256) : "memory");
            }
        }

        // B fragment source in LDS: lane holds column j0+l15;
        // halves = K c*32 + 8*hi + {0..7, 16..23} at padded row stride.
        const _Float16* bl = &btile[jt & 1][(size_t)l15 * LDS_ROW_H + hi * 8];

        v8f acc = {};
#pragma unroll
        for (int c = 0; c < KCHUNKS; ++c) {
            const v8h* bp = (const v8h*)(bl + c * 32);
            v16h b;
            ((v8h*)&b)[0] = bp[0];
            ((v8h*)&b)[1] = bp[2];
            acc = __builtin_amdgcn_wmma_f32_16x16x32_f16(
                false, afrag[c], false, b, (short)0, acc, false, false);
        }

        const bool isDiagTile = (j0 == i0);
        const bool isPosTile  = (j0 == jpos);
#pragma unroll
        for (int r = 0; r < 8; ++r) {
            const int  M        = r + 8 * hi;         // C/D layout: M = r + 8*half
            const bool diagLane = (l15 == M);         // N == M within tile
            const float sim = acc[r];
            float e = __builtin_amdgcn_exp2f(sim * INV_T_LOG2E);
            if (isDiagTile && diagLane) e = 0.0f;     // mask = 1 - eye
            rowpart[r] += e;
            if (isPosTile && diagLane) pos[i0 + M] = sim;
        }

        // Own async issues complete, then block-wide visibility + buffer swap.
        asm volatile("s_wait_asynccnt 0x0" ::: "memory");
        __syncthreads();
    }

    // Reduce row partials across each 16-lane half (N dimension).
#pragma unroll
    for (int r = 0; r < 8; ++r) {
        float v = rowpart[r];
        v += __shfl_xor(v, 1, 32);
        v += __shfl_xor(v, 2, 32);
        v += __shfl_xor(v, 4, 32);
        v += __shfl_xor(v, 8, 32);
        rowpart[r] = v;
    }
#pragma unroll
    for (int r = 0; r < 8; ++r) {
        const int M = r + 8 * hi;
        if (l15 == M)
            rowsum_part[(size_t)seg * NROWS + i0 + M] = rowpart[r];
    }
}

// ---------------------------------------------------------------------------
// Kernel 3: loss = mean_i( log(sum_seg rowsum_part[seg][i]) - 2*pos[i] )
// Fixed-order reduction (deterministic).
// ---------------------------------------------------------------------------
__global__ void infonce_reduce(const float* __restrict__ rowsum_part,
                               const float* __restrict__ pos,
                               float* __restrict__ out) {
    __shared__ float sm[256];
    float acc = 0.f;
    for (int i = threadIdx.x; i < NROWS; i += 256) {
        float den = 0.f;
#pragma unroll
        for (int s = 0; s < JSEG; ++s) den += rowsum_part[(size_t)s * NROWS + i];
        acc += __builtin_amdgcn_logf(den) * LN2F - 2.0f * pos[i];
    }
    sm[threadIdx.x] = acc;
    __syncthreads();
    for (int s = 128; s > 0; s >>= 1) {
        if ((int)threadIdx.x < s) sm[threadIdx.x] += sm[threadIdx.x + s];
        __syncthreads();
    }
    if (threadIdx.x == 0) out[0] = sm[0] / (float)NROWS;
}

// ---------------------------------------------------------------------------
extern "C" void kernel_launch(void* const* d_in, const int* in_sizes, int n_in,
                              void* d_out, int out_size, void* d_ws, size_t ws_size,
                              hipStream_t stream) {
    const float* poly = (const float*)d_in[0];
    const float* cemb = (const float*)d_in[1];

    _Float16* zh     = (_Float16*)d_ws;                                   // 8 MB
    float*    rowsum = (float*)((char*)d_ws + (size_t)NROWS * D_DIM * 2); // 256 KB
    float*    pos    = rowsum + (size_t)JSEG * NROWS;                     // 32 KB

    infonce_normalize<<<NROWS / WAVES_PER_BLOCK, 256, 0, stream>>>(poly, cemb, zh);

    dim3 grid(ROW_BLOCKS, JSEG);
    infonce_gram<<<grid, 256, 0, stream>>>(zh, rowsum, pos);

    infonce_reduce<<<1, 256, 0, stream>>>(rowsum, pos, (float*)d_out);
}
